// AreaSelfAttention_31860067401845
// MI455X (gfx1250) — matrix operand order repaired and checked by reading, not compiled, not run on hardware
//
#include <hip/hip_runtime.h>
#include <hip/hip_bf16.h>

// ---------------------------------------------------------------------------
// Area (8x8 window) self-attention for MI455X / gfx1250, f16 WMMA path.
//   x:(4,256,252,252) f32, Wq/Wk:(32,256), Wv:(256,256), out = gamma*attn(x)+x
// One workgroup = one 8x8 window (4096 windows), 256 threads = 8 wave32.
// ---------------------------------------------------------------------------

typedef __attribute__((ext_vector_type(8)))  _Float16 v8h;
typedef __attribute__((ext_vector_type(16))) _Float16 v16h;
typedef __attribute__((ext_vector_type(8)))  float    v8f;

#define C_IN   256
#define C_R    32
#define H_IN   252
#define W_IN   252
#define AREA   8
#define NPIX   64      // 8*8 pixels per window
#define NWIN   4096    // 4 * 32 * 32

// LDS strides (in _Float16 elements / floats), padded to dodge bank conflicts,
// all multiples of 8 halves so every v8h (16B) access stays aligned.
#define XT_S   264     // Xt[pixel][channel]  64 x 264 halves
#define QT_S   40      // qT/kT[pixel][cr]    64 x 40 halves
#define AT_S   72      // attn[i][j]          64 x 72 halves
#define V_S    72      // V[c][j]            256 x 72 halves
#define S_S    72      // S[i][j]             64 x 72 floats

__device__ __forceinline__ v16h cat16(v8h lo, v8h hi) {
    v16h r;
#pragma unroll
    for (int i = 0; i < 8; ++i) { r[i] = lo[i]; r[i + 8] = hi[i]; }
    return r;
}

// A-operand (16x32 f16, MxK) per ISA layout: lane M = lane&15, g = lane>>4,
// VGPR0-3 hold K = g*8..g*8+7, VGPR4-7 hold K = 16+g*8..16+g*8+7.
// Source is row-major [row][k] with leading dim ld -> two 16B loads.
__device__ __forceinline__ v16h load_a16(const _Float16* Wm, int ld,
                                         int mbase, int kbase, int lane) {
    int row = mbase + (lane & 15);
    int g   = lane >> 4;
    const _Float16* p = Wm + row * ld + kbase + g * 8;
    v8h lo = *(const v8h*)(p);
    v8h hi = *(const v8h*)(p + 16);
    return cat16(lo, hi);
}

// B-operand (32x16 f16, KxN) per ISA layout: lane N = lane&15, g = lane>>4,
// elements 0-7 hold K = g*8..g*8+7, elements 8-15 hold K = 16+g*8..16+g*8+7.
// Source is [n][k] row-major (K contiguous) -> two 16B loads.
__device__ __forceinline__ v16h load_b16(const _Float16* Bm, int ld,
                                         int nbase, int kbase, int lane) {
    int n = nbase + (lane & 15);
    int g = lane >> 4;
    const _Float16* p = Bm + n * ld + kbase + g * 8;
    v8h lo = *(const v8h*)(p);
    v8h hi = *(const v8h*)(p + 16);
    return cat16(lo, hi);
}

__device__ __forceinline__ v8f wmma16(v16h a, v16h b, v8f c) {
    return __builtin_amdgcn_wmma_f32_16x16x32_f16(
        /*neg_a=*/false, a, /*neg_b=*/false, b,
        /*c_mod=*/(short)0, c, /*reuse_a=*/false, /*reuse_b=*/false);
}

// ---------------------------------------------------------------------------
// Prep: convert Wq (8192), Wk (8192), Wv (65536) fp32 -> f16 into workspace.
// ---------------------------------------------------------------------------
__global__ void prep_weights(const float* __restrict__ Wq,
                             const float* __restrict__ Wk,
                             const float* __restrict__ Wv,
                             _Float16* __restrict__ wsh) {
    int gid = blockIdx.x * blockDim.x + threadIdx.x;
    if (gid < 8192)        wsh[gid] = (_Float16)Wq[gid];
    else if (gid < 16384)  wsh[gid] = (_Float16)Wk[gid - 8192];
    else if (gid < 81920)  wsh[gid] = (_Float16)Wv[gid - 16384];
}

// ---------------------------------------------------------------------------
// Main fused kernel.
// ---------------------------------------------------------------------------
__global__ void __launch_bounds__(256, 1)
area_attn_kernel(const float* __restrict__ x,
                 const float* __restrict__ bq,
                 const float* __restrict__ bk,
                 const float* __restrict__ bv,
                 const float* __restrict__ gammap,
                 const _Float16* __restrict__ wsh,
                 float* __restrict__ out) {
    extern __shared__ __align__(16) char smem_raw[];
    _Float16* Xt   = (_Float16*)smem_raw;          // 64*264  = 16896 halves
    _Float16* qT   = Xt  + 64 * XT_S;              // 64*40   =  2560
    _Float16* kT   = qT  + 64 * QT_S;              // 64*40   =  2560
    _Float16* attn = kT  + 64 * QT_S;              // 64*72   =  4608
    _Float16* Vl   = attn + 64 * AT_S;             // 256*72  = 18432
    float*    S    = (float*)(Vl + 256 * V_S);     // 64*72 floats

    const _Float16* Wqh = wsh;
    const _Float16* Wkh = wsh + 8192;
    const _Float16* Wvh = wsh + 16384;

    const int tid  = threadIdx.x;
    const int lane = tid & 31;
    const int wv   = tid >> 5;          // wave id 0..7

    const int n   = blockIdx.x;         // window id
    const int bb  = n >> 10;            // batch
    const int wy  = (n >> 5) & 31;      // window row
    const int wx  = n & 31;             // window col
    const float gamma0 = gammap[0];

    // ---------------- Phase A: stage x slab -> Xt[pixel][channel] (f16) ----
    {
        const int p  = tid & 63;        // pixel within window
        const int c0 = tid >> 6;        // 0..3
        const int dy = p >> 3, dx = p & 7;
        const int gh = wy * AREA + dy;
        const int gw = wx * AREA + dx;
        const bool valid = (gh < H_IN) && (gw < W_IN);
        const float* xb = x + ((size_t)bb * C_IN) * (H_IN * W_IN)
                            + (size_t)gh * W_IN + gw;
#pragma unroll 4
        for (int c = c0; c < C_IN; c += 4) {
            float v = valid ? xb[(size_t)c * (H_IN * W_IN)] : 0.0f;
            Xt[p * XT_S + c] = (_Float16)v;
        }
    }
    __syncthreads();

    // ---------------- Phase B: q,k projections (32 x 64) -------------------
    {
        const int mb = (wv >> 2) * 16;  // 0 or 16  (q/k channel tile)
        const int nb = (wv & 3) * 16;   // pixel tile
        v8f accq = {}; v8f acck = {};
#pragma unroll
        for (int kb = 0; kb < C_IN; kb += 32) {
            v16h bx = load_b16(Xt, XT_S, nb, kb, lane);
            v16h aq = load_a16(Wqh, C_IN, mb, kb, lane);
            accq = wmma16(aq, bx, accq);
            v16h ak = load_a16(Wkh, C_IN, mb, kb, lane);
            acck = wmma16(ak, bx, acck);
        }
#pragma unroll
        for (int r = 0; r < 8; ++r) {
            int c = mb + r + 8 * (lane >> 4);
            int p = nb + (lane & 15);
            qT[p * QT_S + c] = (_Float16)(accq[r] + bq[c]);
            kT[p * QT_S + c] = (_Float16)(acck[r] + bk[c]);
        }
    }
    __syncthreads();

    // ---------------- Phase C: S = q^T k  (64 x 64, K=32) ------------------
    {
#pragma unroll
        for (int t = wv; t < 16; t += 8) {
            const int mb = (t >> 2) * 16;   // i tile
            const int nb = (t & 3) * 16;    // j tile
            v16h a = load_a16(qT, QT_S, mb, 0, lane);   // A = Q^T (i x c)
            v16h b = load_b16(kT, QT_S, nb, 0, lane);   // B = K   (c x j)
            v8f s = {};
            s = wmma16(a, b, s);
#pragma unroll
            for (int r = 0; r < 8; ++r)
                S[(mb + r + 8 * (lane >> 4)) * S_S + nb + (lane & 15)] = s[r];
        }
    }
    __syncthreads();

    // ---------------- Phase D: softmax over j, store attn f16 --------------
    if (tid < 64) {
        const int i = tid;
        float m = -1e30f;
#pragma unroll 8
        for (int j = 0; j < NPIX; ++j) m = fmaxf(m, S[i * S_S + j]);
        float sum = 0.0f;
#pragma unroll 8
        for (int j = 0; j < NPIX; ++j) sum += __expf(S[i * S_S + j] - m);
        float inv = __frcp_rn(sum);
#pragma unroll 8
        for (int j = 0; j < NPIX; ++j)
            attn[i * AT_S + j] = (_Float16)(__expf(S[i * S_S + j] - m) * inv);
    }
    __syncthreads();

    // ---------------- Phase E: V = Wv * X + bv  (256 x 64) -----------------
    {
#pragma unroll
        for (int mt = 0; mt < 2; ++mt) {
            const int mb = (wv * 2 + mt) * 16;
#pragma unroll
            for (int nt = 0; nt < 4; ++nt) {
                const int nb = nt * 16;
                v8f acc = {};
#pragma unroll
                for (int kb = 0; kb < C_IN; kb += 32) {
                    v16h a = load_a16(Wvh, C_IN, mb, kb, lane);
                    v16h b = load_b16(Xt, XT_S, nb, kb, lane);
                    acc = wmma16(a, b, acc);
                }
#pragma unroll
                for (int r = 0; r < 8; ++r) {
                    int c = mb + r + 8 * (lane >> 4);
                    int j = nb + (lane & 15);
                    Vl[c * V_S + j] = (_Float16)(acc[r] + bv[c]);
                }
            }
        }
    }
    __syncthreads();

    // ---------------- Phase F: out = V * attn^T; gamma*out + x -------------
    {
#pragma unroll
        for (int mt = 0; mt < 2; ++mt) {
            const int mb = (wv * 2 + mt) * 16;
#pragma unroll
            for (int nt = 0; nt < 4; ++nt) {
                const int nb = nt * 16;
                v8f acc = {};
#pragma unroll
                for (int kb = 0; kb < NPIX; kb += 32) {
                    v16h a = load_a16(Vl, V_S, mb, kb, lane);   // A = V (c x j)
                    v16h b = load_b16(attn, AT_S, nb, kb, lane);// B = attn^T
                    acc = wmma16(a, b, acc);
                }
#pragma unroll
                for (int r = 0; r < 8; ++r) {
                    int c = mb + r + 8 * (lane >> 4);
                    int p = nb + (lane & 15);
                    int dy = p >> 3, dx = p & 7;
                    int gh = wy * AREA + dy;
                    int gw = wx * AREA + dx;
                    if (gh < H_IN && gw < W_IN) {
                        size_t idx = ((size_t)(bb * C_IN + c) * H_IN + gh) * W_IN + gw;
                        out[idx] = gamma0 * acc[r] + x[idx];
                    }
                }
            }
        }
    }
}

// ---------------------------------------------------------------------------
extern "C" void kernel_launch(void* const* d_in, const int* in_sizes, int n_in,
                              void* d_out, int out_size, void* d_ws, size_t ws_size,
                              hipStream_t stream) {
    const float* x      = (const float*)d_in[0];
    const float* Wq     = (const float*)d_in[1];
    const float* bq     = (const float*)d_in[2];
    const float* Wk     = (const float*)d_in[3];
    const float* bk     = (const float*)d_in[4];
    const float* Wv     = (const float*)d_in[5];
    const float* bv     = (const float*)d_in[6];
    const float* gamma  = (const float*)d_in[7];
    float* out          = (float*)d_out;
    _Float16* wsh       = (_Float16*)d_ws;   // needs 81920*2 = 160 KB

    // f32 -> f16 weight conversion (L2-resident operand pool).
    prep_weights<<<320, 256, 0, stream>>>(Wq, Wk, Wv, wsh);

    // Dynamic LDS: 108544 bytes (> default 64KB cap, raise the limit).
    const size_t smem = (64 * XT_S + 2 * 64 * QT_S + 64 * AT_S + 256 * V_S) * sizeof(_Float16)
                      + 64 * S_S * sizeof(float);
    hipFuncSetAttribute((const void*)area_attn_kernel,
                        hipFuncAttributeMaxDynamicSharedMemorySize, (int)smem);

    area_attn_kernel<<<NWIN, 256, smem, stream>>>(x, bq, bk, bv, gamma, wsh, out);
}
